// GEFM_30863634989620
// MI455X (gfx1250) — matrix operand order repaired and checked by reading, not compiled, not run on hardware
//
#include <hip/hip_runtime.h>
#include <hip/hip_bf16.h>

typedef __bf16 bf16;
typedef __attribute__((ext_vector_type(16))) __bf16 v16bf;
typedef __attribute__((ext_vector_type(8)))  __bf16 v8bf;
typedef __attribute__((ext_vector_type(8)))  float   v8f;

#define NPIX  4096
#define CDIM  64
#define IMGW  64
#define PPAD  40   // bf16 elements per LDS row of P tile (row stride 80B, 16B aligned)

static __device__ inline v16bf cat8(v8bf lo, v8bf hi) {
  return __builtin_shufflevector(lo, hi, 0,1,2,3,4,5,6,7,8,9,10,11,12,13,14,15);
}
static __device__ inline int iclamp(int v, int lo, int hi) {
  return v < lo ? lo : (v > hi ? hi : v);
}

// ------------------------------- f32 NCHW -> bf16 NHWC convert (x and y fused)
__global__ void cvt_nhwc_kernel(const float* __restrict__ x, const float* __restrict__ y,
                                bf16* __restrict__ out, int n2) {
  int i = blockIdx.x * blockDim.x + threadIdx.x;   // i indexes NHWC output (xb then yb)
  if (i >= n2) return;
  const int c  = i & 63;
  const int p  = (i >> 6) & (NPIX - 1);
  const int bb = i >> 18;                           // 0..7: 0-3 -> x, 4-7 -> y
  const float* src = (bb < 4) ? x : y;
  out[i] = (bf16)src[((size_t)((bb & 3) * CDIM + c)) * NPIX + p];
}

// ------------------------------------- pack weights into WMMA A-fragment order
// Wp flat index = ((((coT*(Cin/32)+chunk)*9 + tap)*32)+lane)*16 + e
__global__ void pack_w_kernel(const float* __restrict__ W, bf16* __restrict__ Wp, int Cin) {
  const int i = blockIdx.x * blockDim.x + threadIdx.x;
  const int total = 4 * (Cin / 32) * 9 * 32 * 16;
  if (i >= total) return;
  const int e     = i & 15;
  const int lane  = (i >> 4) & 31;
  const int tap   = (i >> 9) % 9;
  const int chunk = (i / (512 * 9)) % (Cin / 32);
  const int coT   = i / (512 * 9 * (Cin / 32));
  const int g = lane >> 4, ln = lane & 15;
  const int co = coT * 16 + ln;
  const int k  = e + 16 * g;
  const int ci = chunk * 32 + k;
  Wp[i] = (bf16)W[((size_t)(co * Cin + ci) * 3 + tap / 3) * 3 + (tap % 3)];
}

// --------------------------------------------------- Conv3x3 + BN + ReLU, WMMA
// Core: 16 cout x 64 pixels (one full image row) per wave.
// Each weight A-fragment is reused by 4 pixel subtiles (4 WMMAs per 2 A-loads).
static __device__ __forceinline__
void conv_core(const bf16* __restrict__ src0, const bf16* __restrict__ src1,
               const bf16* __restrict__ Wp, const float* __restrict__ sc,
               const float* __restrict__ bi, bf16* outb, int outb_nhwc,
               float* outf, int Cin, int b, int row, int coT, int lane)
{
  const int g  = lane >> 4;
  const int ln = lane & 15;
  const int nchunk = Cin >> 5;
  const v8bf zero = {};
  v8f acc[4] = {};

  for (int chunk = 0; chunk < nchunk; ++chunk) {
    const int cl = (chunk & 1) * 32;                 // channel offset within tensor
    const bf16* src = (chunk < 2) ? src0 : src1;     // nchunk==2 -> always src0
    const bf16* wbase = Wp + (size_t)((coT * nchunk + chunk) * 9) * 512;
    #pragma unroll
    for (int ky = 0; ky < 3; ++ky) {
      const int hh = row + ky - 1;
      const bool rowok = (hh >= 0) && (hh < IMGW);
      const int hcl = iclamp(hh, 0, IMGW - 1);
      #pragma unroll
      for (int kx = 0; kx < 3; ++kx) {
        // A fragment: packed weights, 2 contiguous b128 per lane, reused 4x
        const bf16* wp = wbase + (size_t)((ky * 3 + kx) * 32 + lane) * 16;
        const v16bf a = cat8(*(const v8bf*)wp, *(const v8bf*)(wp + 8));
        #pragma unroll
        for (int s = 0; s < 4; ++s) {
          const int ww = 16 * s + ln + kx - 1;
          const bool ok = rowok && (ww >= 0) && (ww < IMGW);
          const int p = hcl * IMGW + iclamp(ww, 0, IMGW - 1);
          const bf16* bp = src + (size_t)p * CDIM + cl + 16 * g;
          v8bf blo = *(const v8bf*)bp;
          v8bf bhi = *(const v8bf*)(bp + 8);
          blo = ok ? blo : zero;
          bhi = ok ? bhi : zero;
          acc[s] = __builtin_amdgcn_wmma_f32_16x16x32_bf16(false, a, false, cat8(blo, bhi),
                                                           (short)0, acc[s], false, false);
        }
      }
    }
  }
  // Epilogue: folded BN + ReLU
  #pragma unroll
  for (int r = 0; r < 8; ++r) {
    const int co = coT * 16 + r + 8 * g;
    const float scl = sc[co], bia = bi[co];
    #pragma unroll
    for (int s = 0; s < 4; ++s) {
      const int pix = row * IMGW + 16 * s + ln;
      float v = acc[s][r] * scl + bia;
      v = v > 0.f ? v : 0.f;
      if (outf) outf[(size_t)b * CDIM * NPIX + (size_t)co * NPIX + pix] = v;
      if (outb) {
        const size_t idx = outb_nhwc ? (size_t)b * NPIX * CDIM + (size_t)pix * CDIM + co
                                     : (size_t)b * CDIM * NPIX + (size_t)co * NPIX + pix;
        outb[idx] = (bf16)v;
      }
    }
  }
}

// 128-channel-input conv (Q and final SR): grid (64 rows, 4 coT, 4 batch)
__global__ __launch_bounds__(32)
void conv128_kernel(const bf16* __restrict__ in0, const bf16* __restrict__ in1,
                    const bf16* __restrict__ Wp, const float* __restrict__ sc,
                    const float* __restrict__ bi, bf16* outb, int outb_nhwc, float* outf)
{
  const int b = blockIdx.z;
  conv_core(in0 + (size_t)b * NPIX * CDIM, in1 + (size_t)b * NPIX * CDIM,
            Wp, sc, bi, outb, outb_nhwc, outf, 128, b, blockIdx.x, blockIdx.y, threadIdx.x);
}

// All four 64-channel K/V convs fused in one launch: grid (64, 4, 16) z = job*4 + b
struct ConvJobs {
  const bf16* in[4]; const bf16* wp[4]; const float* sc[4]; const float* bi[4];
  bf16* out[4]; int nhwc[4];
};
__global__ __launch_bounds__(32)
void conv64x4_kernel(ConvJobs J)
{
  const int b = blockIdx.z & 3;
  const int j = blockIdx.z >> 2;
  conv_core(J.in[j] + (size_t)b * NPIX * CDIM, nullptr,
            J.wp[j], J.sc[j], J.bi[j], J.out[j], J.nhwc[j], nullptr,
            64, b, blockIdx.x, blockIdx.y, threadIdx.x);
}

// --------------------------------------------- Flash attention + gamma*out+res
// refine[c,m] = sum_q softmax_q(K[:,m].Q[:,q]) * V[c,q]
// 32 query rows (two 16-row m-tiles) per wave; Q/V fragments shared across m-tiles.
// Both streams (RGB/INF) fused: blockIdx.z selects per-stream K/V/res/gamma/out.
struct AttnJobs {
  const bf16* K[2]; const bf16* V[2]; const float* res[2]; const float* gam[2]; bf16* out[2];
  const bf16* Q;
};
__global__ __launch_bounds__(32)
void flash_attn_wmma_kernel(AttnJobs J)
{
  __shared__ bf16 Ps[32 * PPAD];   // probability tiles [m_local 0..31][q 0..31]

  const int lane = threadIdx.x;
  const int g = lane >> 4, ln = lane & 15;
  const int m0 = blockIdx.x * 32;
  const int b  = blockIdx.y;
  const int st = blockIdx.z;

  const bf16*  Kn = J.K[st]   + (size_t)b * NPIX * CDIM;
  const bf16*  Qn = J.Q       + (size_t)b * NPIX * CDIM;
  const bf16*  Vm = J.V[st]   + (size_t)b * CDIM * NPIX;
  const float* rs = J.res[st] + (size_t)b * CDIM * NPIX;
  bf16*        op = J.out[st] + (size_t)b * NPIX * CDIM;
  const float* gamma = J.gam[st];

  // Query A fragments A[m][c] = K[m][c] (NHWC), two m-tiles, c split 0..31/32..63
  const bf16* kp0 = Kn + (size_t)(m0 + ln) * CDIM;
  const bf16* kp1 = Kn + (size_t)(m0 + 16 + ln) * CDIM;
  const v16bf aK00 = cat8(*(const v8bf*)(kp0 + 16 * g),      *(const v8bf*)(kp0 + 8 + 16 * g));
  const v16bf aK01 = cat8(*(const v8bf*)(kp0 + 32 + 16 * g), *(const v8bf*)(kp0 + 40 + 16 * g));
  const v16bf aK10 = cat8(*(const v8bf*)(kp1 + 16 * g),      *(const v8bf*)(kp1 + 8 + 16 * g));
  const v16bf aK11 = cat8(*(const v8bf*)(kp1 + 32 + 16 * g), *(const v8bf*)(kp1 + 40 + 16 * g));

  v8f o0[4] = {}, o1[4] = {};
  float mrow0[8], lrow0[8], mrow1[8], lrow1[8];
  #pragma unroll
  for (int r = 0; r < 8; ++r) {
    mrow0[r] = -1e30f; lrow0[r] = 0.f;
    mrow1[r] = -1e30f; lrow1[r] = 0.f;
  }

  for (int kb = 0; kb < NPIX; kb += 32) {
    // Prefetch next key block (speculative; harmless past the end)
    __builtin_prefetch(Qn + (size_t)(kb + 32 + ln) * CDIM, 0, 3);
    __builtin_prefetch(Qn + (size_t)(kb + 48 + ln) * CDIM, 0, 3);
    #pragma unroll
    for (int ct = 0; ct < 4; ++ct)
      __builtin_prefetch(Vm + (size_t)(ct * 16 + ln) * NPIX + kb + 32, 0, 3);

    // Shared Q B-fragments for this 32-key block (two q-tiles x two c-halves)
    const bf16* q0 = Qn + (size_t)(kb + ln) * CDIM + 16 * g;
    const bf16* q1 = Qn + (size_t)(kb + 16 + ln) * CDIM + 16 * g;
    const v16bf bQ00 = cat8(*(const v8bf*)q0,        *(const v8bf*)(q0 + 8));
    const v16bf bQ01 = cat8(*(const v8bf*)(q0 + 32), *(const v8bf*)(q0 + 40));
    const v16bf bQ10 = cat8(*(const v8bf*)q1,        *(const v8bf*)(q1 + 8));
    const v16bf bQ11 = cat8(*(const v8bf*)(q1 + 32), *(const v8bf*)(q1 + 40));

    // Logits: 4 s-tiles (2 m-tiles x 2 q-tiles), 2 WMMAs each (c=64)
    v8f s00 = {}, s01 = {}, s10 = {}, s11 = {};
    s00 = __builtin_amdgcn_wmma_f32_16x16x32_bf16(false, aK00, false, bQ00, (short)0, s00, false, false);
    s00 = __builtin_amdgcn_wmma_f32_16x16x32_bf16(false, aK01, false, bQ01, (short)0, s00, false, false);
    s01 = __builtin_amdgcn_wmma_f32_16x16x32_bf16(false, aK00, false, bQ10, (short)0, s01, false, false);
    s01 = __builtin_amdgcn_wmma_f32_16x16x32_bf16(false, aK01, false, bQ11, (short)0, s01, false, false);
    s10 = __builtin_amdgcn_wmma_f32_16x16x32_bf16(false, aK10, false, bQ00, (short)0, s10, false, false);
    s10 = __builtin_amdgcn_wmma_f32_16x16x32_bf16(false, aK11, false, bQ01, (short)0, s10, false, false);
    s11 = __builtin_amdgcn_wmma_f32_16x16x32_bf16(false, aK10, false, bQ10, (short)0, s11, false, false);
    s11 = __builtin_amdgcn_wmma_f32_16x16x32_bf16(false, aK11, false, bQ11, (short)0, s11, false, false);

    // Online softmax, m-tile 0
    #pragma unroll
    for (int r = 0; r < 8; ++r) {
      const float v0 = s00[r], v1 = s01[r];
      float mx = fmaxf(v0, v1);
      #pragma unroll
      for (int sh = 1; sh < 16; sh <<= 1) mx = fmaxf(mx, __shfl_xor(mx, sh, 32));
      const float newm  = fmaxf(mrow0[r], mx);
      const float scale = __expf(mrow0[r] - newm);
      const float p0 = __expf(v0 - newm);
      const float p1 = __expf(v1 - newm);
      float ps = p0 + p1;
      #pragma unroll
      for (int sh = 1; sh < 16; sh <<= 1) ps += __shfl_xor(ps, sh, 32);
      lrow0[r] = lrow0[r] * scale + ps;
      mrow0[r] = newm;
      #pragma unroll
      for (int ct = 0; ct < 4; ++ct) o0[ct][r] *= scale;
      const int ml = r + 8 * g;
      Ps[ml * PPAD + ln]      = (bf16)p0;
      Ps[ml * PPAD + 16 + ln] = (bf16)p1;
    }
    // Online softmax, m-tile 1
    #pragma unroll
    for (int r = 0; r < 8; ++r) {
      const float v0 = s10[r], v1 = s11[r];
      float mx = fmaxf(v0, v1);
      #pragma unroll
      for (int sh = 1; sh < 16; sh <<= 1) mx = fmaxf(mx, __shfl_xor(mx, sh, 32));
      const float newm  = fmaxf(mrow1[r], mx);
      const float scale = __expf(mrow1[r] - newm);
      const float p0 = __expf(v0 - newm);
      const float p1 = __expf(v1 - newm);
      float ps = p0 + p1;
      #pragma unroll
      for (int sh = 1; sh < 16; sh <<= 1) ps += __shfl_xor(ps, sh, 32);
      lrow1[r] = lrow1[r] * scale + ps;
      mrow1[r] = newm;
      #pragma unroll
      for (int ct = 0; ct < 4; ++ct) o1[ct][r] *= scale;
      const int ml = 16 + r + 8 * g;
      Ps[ml * PPAD + ln]      = (bf16)p0;
      Ps[ml * PPAD + 16 + ln] = (bf16)p1;
    }
    asm volatile("s_wait_dscnt 0" ::: "memory");

    // P A-fragments (C/D layout -> A layout via LDS bounce), one per m-tile
    const v16bf aP0 = cat8(*(const v8bf*)&Ps[ln * PPAD + 8 * g],
                           *(const v8bf*)&Ps[ln * PPAD + 16 + 8 * g]);
    const v16bf aP1 = cat8(*(const v8bf*)&Ps[(16 + ln) * PPAD + 8 * g],
                           *(const v8bf*)&Ps[(16 + ln) * PPAD + 16 + 8 * g]);

    // O[m,c] += P[m,q] * V^T[q,c]; V fragments shared across both m-tiles
    #pragma unroll
    for (int ct = 0; ct < 4; ++ct) {
      const bf16* vp = Vm + (size_t)(ct * 16 + ln) * NPIX + kb + 8 * g;
      const v16bf bV = cat8(*(const v8bf*)vp, *(const v8bf*)(vp + 16));
      o0[ct] = __builtin_amdgcn_wmma_f32_16x16x32_bf16(false, aP0, false, bV, (short)0, o0[ct], false, false);
      o1[ct] = __builtin_amdgcn_wmma_f32_16x16x32_bf16(false, aP1, false, bV, (short)0, o1[ct], false, false);
    }
  }

  // Epilogue: normalize, gamma * refine + residual, store NHWC bf16
  const float gm = gamma[0];
  #pragma unroll
  for (int r = 0; r < 8; ++r) {
    const float inv0 = 1.f / lrow0[r];
    const float inv1 = 1.f / lrow1[r];
    const int ma = m0 + r + 8 * g;
    const int mb = m0 + 16 + r + 8 * g;
    #pragma unroll
    for (int ct = 0; ct < 4; ++ct) {
      const int c = ct * 16 + ln;
      op[(size_t)ma * CDIM + c] = (bf16)(gm * o0[ct][r] * inv0 + rs[(size_t)c * NPIX + ma]);
      op[(size_t)mb * CDIM + c] = (bf16)(gm * o1[ct][r] * inv1 + rs[(size_t)c * NPIX + mb]);
    }
  }
}

// ---------------------------------------------------------------------- launch
extern "C" void kernel_launch(void* const* d_in, const int* in_sizes, int n_in,
                              void* d_out, int out_size, void* d_ws, size_t ws_size,
                              hipStream_t stream) {
  (void)in_sizes; (void)n_in; (void)out_size; (void)ws_size;
  const float* x   = (const float*)d_in[0];
  const float* y   = (const float*)d_in[1];
  const float* Wq  = (const float*)d_in[2];   const float* sq  = (const float*)d_in[3];  const float* bq  = (const float*)d_in[4];
  const float* Wrk = (const float*)d_in[5];   const float* srk = (const float*)d_in[6];  const float* brk = (const float*)d_in[7];
  const float* Wrv = (const float*)d_in[8];   const float* srv = (const float*)d_in[9];  const float* brv = (const float*)d_in[10];
  const float* Wik = (const float*)d_in[11];  const float* sik = (const float*)d_in[12]; const float* bik = (const float*)d_in[13];
  const float* Wiv = (const float*)d_in[14];  const float* siv = (const float*)d_in[15]; const float* biv = (const float*)d_in[16];
  const float* Wsr = (const float*)d_in[17];  const float* ssr = (const float*)d_in[18]; const float* bsr = (const float*)d_in[19];
  const float* g1  = (const float*)d_in[20];
  const float* g2  = (const float*)d_in[21];

  const size_t TSZ   = (size_t)4 * CDIM * NPIX;   // elements per [B,64,4096] tensor
  const int PACK64   = 4 * 2 * 9 * 512;           // 36864 packed bf16, Cin=64
  const int PACK128  = 4 * 4 * 9 * 512;           // 73728 packed bf16, Cin=128
  bf16* xb   = (bf16*)d_ws;           // NHWC
  bf16* yb   = xb  + TSZ;             // NHWC
  bf16* Qb   = yb  + TSZ;             // NHWC
  bf16* Krb  = Qb  + TSZ;             // NHWC
  bf16* Kib  = Krb + TSZ;             // NHWC
  bf16* Vrb  = Kib + TSZ;             // NCHW
  bf16* Vib  = Vrb + TSZ;             // NCHW
  bf16* Rr   = Vib + TSZ;             // NHWC
  bf16* Ri   = Rr  + TSZ;             // NHWC
  bf16* WqP  = Ri  + TSZ;
  bf16* WrkP = WqP  + PACK128;
  bf16* WrvP = WrkP + PACK64;
  bf16* WikP = WrvP + PACK64;
  bf16* WivP = WikP + PACK64;
  bf16* WsrP = WivP + PACK64;

  const int n2 = (int)(2 * TSZ);
  cvt_nhwc_kernel<<<(n2 + 255) / 256, 256, 0, stream>>>(x, y, xb, n2);

  pack_w_kernel<<<(PACK128 + 255) / 256, 256, 0, stream>>>(Wq,  WqP,  128);
  pack_w_kernel<<<(PACK64  + 255) / 256, 256, 0, stream>>>(Wrk, WrkP, 64);
  pack_w_kernel<<<(PACK64  + 255) / 256, 256, 0, stream>>>(Wrv, WrvP, 64);
  pack_w_kernel<<<(PACK64  + 255) / 256, 256, 0, stream>>>(Wik, WikP, 64);
  pack_w_kernel<<<(PACK64  + 255) / 256, 256, 0, stream>>>(Wiv, WivP, 64);
  pack_w_kernel<<<(PACK128 + 255) / 256, 256, 0, stream>>>(Wsr, WsrP, 128);

  dim3 cblk(32);
  // Q = CBR(concat(x,y)) -> NHWC bf16
  conv128_kernel<<<dim3(64, 4, 4), cblk, 0, stream>>>(xb, yb, WqP, sq, bq, Qb, 1, nullptr);

  // Four K/V convs fused into one launch (z = job*4 + batch)
  ConvJobs cj;
  cj.in[0] = xb; cj.wp[0] = WrkP; cj.sc[0] = srk; cj.bi[0] = brk; cj.out[0] = Krb; cj.nhwc[0] = 1;
  cj.in[1] = xb; cj.wp[1] = WrvP; cj.sc[1] = srv; cj.bi[1] = brv; cj.out[1] = Vrb; cj.nhwc[1] = 0;
  cj.in[2] = yb; cj.wp[2] = WikP; cj.sc[2] = sik; cj.bi[2] = bik; cj.out[2] = Kib; cj.nhwc[2] = 1;
  cj.in[3] = yb; cj.wp[3] = WivP; cj.sc[3] = siv; cj.bi[3] = biv; cj.out[3] = Vib; cj.nhwc[3] = 0;
  conv64x4_kernel<<<dim3(64, 4, 16), cblk, 0, stream>>>(cj);

  // Both attention streams fused into one launch (z = stream)
  AttnJobs aj;
  aj.Q = Qb;
  aj.K[0] = Krb; aj.V[0] = Vrb; aj.res[0] = y; aj.gam[0] = g1; aj.out[0] = Rr;
  aj.K[1] = Kib; aj.V[1] = Vib; aj.res[1] = x; aj.gam[1] = g2; aj.out[1] = Ri;
  flash_attn_wmma_kernel<<<dim3(NPIX / 32, 4, 2), cblk, 0, stream>>>(aj);

  // Final CBR(concat(RGB_refine, INF_refine)) -> f32 NCHW output
  conv128_kernel<<<dim3(64, 4, 4), cblk, 0, stream>>>(Rr, Ri, WsrP, ssr, bsr, nullptr, 0, (float*)d_out);
}
